// TransformerDecoderLayer_45672682226582
// MI455X (gfx1250) — matrix operand order repaired and checked by reading, not compiled, not run on hardware
//
#include <hip/hip_runtime.h>
#include <hip/hip_bf16.h>

typedef __bf16 bf16;
typedef bf16  v16bf __attribute__((ext_vector_type(16)));
typedef bf16  v8bf  __attribute__((ext_vector_type(8)));
typedef float v8f   __attribute__((ext_vector_type(8)));

#define D_MODEL 1024
#define D_FF    4096
#define NHEAD   16
#define HDIM    64
#define BATCH   2
#define SEQ     2048
#define MTOT    (BATCH * SEQ)   // 4096 rows

static __device__ __forceinline__ v16bf make_frag(const bf16* p0, const bf16* p1) {
    v8bf lo = *(const v8bf*)p0;
    v8bf hi = *(const v8bf*)p1;
    return __builtin_shufflevector(lo, hi, 0,1,2,3,4,5,6,7,8,9,10,11,12,13,14,15);
}

// ---------------------------------------------------------------------------
// GEMM: Y[M,N] = X[M,K] @ W[K,N] + bias (+optional ReLU), bf16 WMMA, f32 acc.
// Block tile 256x128, BK=32, 256 threads = 8 waves (4x2), wave tile 64x64.
// Double-buffered LDS; next tile's global loads issued before current compute.
// ---------------------------------------------------------------------------
#define GSTR 40   // LDS k-stride (32 + 8 pad), keeps 16B alignment of frag reads

__global__ __launch_bounds__(256) void gemm_bias_kernel(
    const float* __restrict__ X, const float* __restrict__ W,
    const float* __restrict__ bias, float* __restrict__ Y,
    int M, int N, int K, int relu)
{
    __shared__ bf16 sa[2][256 * GSTR];   // [buf][m][k]
    __shared__ bf16 sb[2][128 * GSTR];   // [buf][n][k]  (W transposed)

    const int tid  = threadIdx.x;
    const int lane = tid & 31, wid = tid >> 5;
    const int wm = wid >> 1, wn = wid & 1;     // 4 x 2 wave grid
    const int g = lane >> 4, ln = lane & 15;
    const int row0 = blockIdx.y * 256;
    const int col0 = blockIdx.x * 128;

    // per-thread staging registers for one K-slab
    float4 a_reg[8];   // A: 256x32 floats = 2048 float4 / 256 thr
    float  b_reg[16];  // B: 32x128 floats / 256 thr

    // A addressing: idx = tid + i*256 -> r = idx>>3, c4 = (idx&7)<<2
    // B addressing: idx = tid + i*256 -> kk = idx>>7, n = idx&127

    auto load_regs = [&](int k0) {
        #pragma unroll
        for (int i = 0; i < 8; ++i) {
            int idx = tid + i * 256;
            int r = idx >> 3, c4 = (idx & 7) << 2;
            a_reg[i] = *(const float4*)&X[(size_t)(row0 + r) * K + k0 + c4];
        }
        #pragma unroll
        for (int i = 0; i < 16; ++i) {
            int idx = tid + i * 256;
            int kk = idx >> 7, n = idx & 127;
            b_reg[i] = W[(size_t)(k0 + kk) * N + col0 + n];
        }
    };
    auto store_lds = [&](int buf) {
        #pragma unroll
        for (int i = 0; i < 8; ++i) {
            int idx = tid + i * 256;
            int r = idx >> 3, c4 = (idx & 7) << 2;
            bf16* d = &sa[buf][r * GSTR + c4];
            d[0] = (bf16)a_reg[i].x; d[1] = (bf16)a_reg[i].y;
            d[2] = (bf16)a_reg[i].z; d[3] = (bf16)a_reg[i].w;
        }
        #pragma unroll
        for (int i = 0; i < 16; ++i) {
            int idx = tid + i * 256;
            int kk = idx >> 7, n = idx & 127;
            sb[buf][n * GSTR + kk] = (bf16)b_reg[i];
        }
    };

    v8f acc[4][4] = {};

    const int nk = K >> 5;
    load_regs(0);
    store_lds(0);
    __syncthreads();

    for (int kt = 0; kt < nk; ++kt) {
        if (kt + 1 < nk) load_regs((kt + 1) << 5);   // overlap with compute

        const int buf = kt & 1;
        const bf16* sA = sa[buf];
        const bf16* sB = sb[buf];

        v16bf af[4];
        #pragma unroll
        for (int mt = 0; mt < 4; ++mt) {
            const bf16* base = &sA[(wm * 64 + mt * 16 + ln) * GSTR];
            af[mt] = make_frag(base + g * 8, base + 16 + g * 8);
        }
        #pragma unroll
        for (int nt = 0; nt < 4; ++nt) {
            const bf16* bb = &sB[(wn * 64 + nt * 16 + ln) * GSTR];
            v16bf bfv = make_frag(bb + g * 8, bb + 16 + g * 8);
            #pragma unroll
            for (int mt = 0; mt < 4; ++mt)
                acc[mt][nt] = __builtin_amdgcn_wmma_f32_16x16x32_bf16(
                    false, af[mt], false, bfv, (short)0, acc[mt][nt], false, false);
        }

        if (kt + 1 < nk) {
            store_lds((kt + 1) & 1);   // other buffer than the one just read
            __syncthreads();
        }
    }

    // Epilogue: bias (+ReLU), store fp32
    #pragma unroll
    for (int mt = 0; mt < 4; ++mt) {
        #pragma unroll
        for (int nt = 0; nt < 4; ++nt) {
            int n = col0 + wn * 64 + nt * 16 + ln;
            float bv = bias[n];
            int mbase = row0 + wm * 64 + mt * 16 + g * 8;
            #pragma unroll
            for (int r = 0; r < 8; ++r) {
                float v = acc[mt][nt][r] + bv;
                if (relu) v = fmaxf(v, 0.f);
                Y[(size_t)(mbase + r) * N + n] = v;
            }
        }
    }
}

// ---------------------------------------------------------------------------
// Flash attention: one block per (q-block of 64, head, batch). 128 thr = 4 waves,
// each wave owns 16 query rows. Streaming softmax, bf16 WMMA for QK^T and P*V.
// ---------------------------------------------------------------------------
#define KSTR 72

__global__ __launch_bounds__(128) void flash_attn_kernel(
    const float* __restrict__ Q, const float* __restrict__ Kp,
    const float* __restrict__ V, float* __restrict__ Out, int causal)
{
    __shared__ bf16 qs[64 * KSTR];       // [qrow][hd]
    __shared__ bf16 kt[64 * KSTR];       // [key][hd]
    __shared__ bf16 vt[64 * KSTR];       // [hd][key] (transposed)
    __shared__ bf16 ps[4 * 16 * KSTR];   // per-wave P scratch [m][key]

    const int qb = blockIdx.x, h = blockIdx.y, b = blockIdx.z;
    const size_t bo = (size_t)b * SEQ * D_MODEL + (size_t)h * HDIM;
    const int tid = threadIdx.x, lane = tid & 31, wid = tid >> 5;
    const int g = lane >> 4, ln = lane & 15;

    // Stage Q block 64x64
    #pragma unroll
    for (int i = 0; i < 8; ++i) {
        int idx = tid + i * 128;             // 0..1023 float4 slots
        int r = idx >> 4, c4 = (idx & 15) << 2;
        const float4 f = *(const float4*)&Q[bo + (size_t)(qb * 64 + r) * D_MODEL + c4];
        bf16* d = &qs[r * KSTR + c4];
        d[0] = (bf16)f.x; d[1] = (bf16)f.y; d[2] = (bf16)f.z; d[3] = (bf16)f.w;
    }
    __syncthreads();

    v16bf qf[2];
    #pragma unroll
    for (int kk = 0; kk < 2; ++kk) {
        const bf16* base = &qs[(wid * 16 + ln) * KSTR + kk * 32];
        qf[kk] = make_frag(base + g * 8, base + 16 + g * 8);
    }

    float mprev[8], lsum[8];
    v8f o[4] = {};
    #pragma unroll
    for (int r = 0; r < 8; ++r) { mprev[r] = -1e30f; lsum[r] = 0.f; }

    const int nkb = causal ? (qb + 1) : (SEQ / 64);
    for (int kb = 0; kb < nkb; ++kb) {
        // Stage K tile [key][hd]
        #pragma unroll
        for (int i = 0; i < 8; ++i) {
            int idx = tid + i * 128;
            int r = idx >> 4, c4 = (idx & 15) << 2;
            const float4 f = *(const float4*)&Kp[bo + (size_t)(kb * 64 + r) * D_MODEL + c4];
            bf16* d = &kt[r * KSTR + c4];
            d[0] = (bf16)f.x; d[1] = (bf16)f.y; d[2] = (bf16)f.z; d[3] = (bf16)f.w;
        }
        // Stage V transposed [hd][key]
        #pragma unroll
        for (int i = 0; i < 8; ++i) {
            int idx = tid + i * 128;
            int r = idx >> 4, c4 = (idx & 15) << 2;
            const float4 f = *(const float4*)&V[bo + (size_t)(kb * 64 + r) * D_MODEL + c4];
            vt[(c4 + 0) * KSTR + r] = (bf16)f.x;
            vt[(c4 + 1) * KSTR + r] = (bf16)f.y;
            vt[(c4 + 2) * KSTR + r] = (bf16)f.z;
            vt[(c4 + 3) * KSTR + r] = (bf16)f.w;
        }
        __syncthreads();

        // Scores: S[16 x 64] = Q(16x64) @ K^T, 4 tiles of 16x16
        v8f st[4];
        #pragma unroll
        for (int nt = 0; nt < 4; ++nt) {
            v8f s = {};
            #pragma unroll
            for (int kk = 0; kk < 2; ++kk) {
                const bf16* bb = &kt[(nt * 16 + ln) * KSTR + kk * 32];
                v16bf kf = make_frag(bb + g * 8, bb + 16 + g * 8);
                s = __builtin_amdgcn_wmma_f32_16x16x32_bf16(
                        false, qf[kk], false, kf, (short)0, s, false, false);
            }
            st[nt] = s;
        }

        // scale + causal mask + per-lane row max
        const bool diag = causal && (kb == qb);
        float rmax[8];
        #pragma unroll
        for (int r = 0; r < 8; ++r) rmax[r] = -1e30f;
        #pragma unroll
        for (int nt = 0; nt < 4; ++nt) {
            #pragma unroll
            for (int r = 0; r < 8; ++r) {
                float v = st[nt][r] * 0.125f;   // 1/sqrt(64)
                if (diag) {
                    int qg = wid * 16 + g * 8 + r;
                    int kg = nt * 16 + ln;
                    if (kg > qg) v = -1e30f;
                }
                st[nt][r] = v;
                rmax[r] = fmaxf(rmax[r], v);
            }
        }

        // streaming softmax update
        #pragma unroll
        for (int r = 0; r < 8; ++r) {
            float m = rmax[r];
            #pragma unroll
            for (int msk = 1; msk < 16; msk <<= 1)
                m = fmaxf(m, __shfl_xor(m, msk, 32));
            float mnew  = fmaxf(mprev[r], m);
            float alpha = __expf(mprev[r] - mnew);
            float psum = 0.f;
            #pragma unroll
            for (int nt = 0; nt < 4; ++nt) {
                float p = __expf(st[nt][r] - mnew);
                st[nt][r] = p;
                psum += p;
            }
            #pragma unroll
            for (int msk = 1; msk < 16; msk <<= 1)
                psum += __shfl_xor(psum, msk, 32);
            lsum[r] = lsum[r] * alpha + psum;
            mprev[r] = mnew;
            #pragma unroll
            for (int nt = 0; nt < 4; ++nt) o[nt][r] *= alpha;
        }

        // Re-stripe P (C-layout -> A-layout) through per-wave LDS scratch
        bf16* pw = &ps[wid * 16 * KSTR];
        #pragma unroll
        for (int nt = 0; nt < 4; ++nt)
            #pragma unroll
            for (int r = 0; r < 8; ++r)
                pw[(g * 8 + r) * KSTR + nt * 16 + ln] = (bf16)st[nt][r];
        __builtin_amdgcn_wave_barrier();   // in-wave DS ordering for reload

        v16bf pf[2];
        #pragma unroll
        for (int kk = 0; kk < 2; ++kk) {
            const bf16* base = &pw[ln * KSTR + kk * 32];
            pf[kk] = make_frag(base + g * 8, base + 16 + g * 8);
        }
        // O += P(16x64) @ V(64x64)
        #pragma unroll
        for (int nt = 0; nt < 4; ++nt) {
            #pragma unroll
            for (int kk = 0; kk < 2; ++kk) {
                const bf16* bb = &vt[(nt * 16 + ln) * KSTR + kk * 32];
                v16bf vf = make_frag(bb + g * 8, bb + 16 + g * 8);
                o[nt] = __builtin_amdgcn_wmma_f32_16x16x32_bf16(
                            false, pf[kk], false, vf, (short)0, o[nt], false, false);
            }
        }
        __syncthreads();   // protect kt/vt before next stage
    }

    // finalize: O /= l, write fp32
    #pragma unroll
    for (int nt = 0; nt < 4; ++nt) {
        int col = nt * 16 + ln;
        #pragma unroll
        for (int r = 0; r < 8; ++r) {
            int rowl = wid * 16 + g * 8 + r;
            Out[bo + (size_t)(qb * 64 + rowl) * D_MODEL + col] = o[nt][r] / lsum[r];
        }
    }
}

// ---------------------------------------------------------------------------
// out = LayerNorm(resid + y) * gamma + beta ; one block per row, D=1024.
// ---------------------------------------------------------------------------
__global__ __launch_bounds__(256) void add_ln_kernel(
    const float* __restrict__ resid, const float* __restrict__ y,
    const float* __restrict__ gam, const float* __restrict__ bet,
    float* __restrict__ out)
{
    const int row = blockIdx.x, tid = threadIdx.x;
    const float* r0 = resid + (size_t)row * D_MODEL;
    const float* y0 = y + (size_t)row * D_MODEL;

    float v[4], s = 0.f, ss = 0.f;
    #pragma unroll
    for (int i = 0; i < 4; ++i) {
        int c = tid + i * 256;
        float t = r0[c] + y0[c];
        v[i] = t; s += t; ss += t * t;
    }
    #pragma unroll
    for (int m = 16; m >= 1; m >>= 1) {
        s  += __shfl_xor(s, m, 32);
        ss += __shfl_xor(ss, m, 32);
    }
    __shared__ float sh[16];
    int lane = tid & 31, wid = tid >> 5;
    if (lane == 0) { sh[wid] = s; sh[8 + wid] = ss; }
    __syncthreads();
    if (tid == 0) {
        float a = 0.f, b2 = 0.f;
        for (int i = 0; i < 8; ++i) { a += sh[i]; b2 += sh[8 + i]; }
        sh[0] = a; sh[8] = b2;
    }
    __syncthreads();
    s = sh[0]; ss = sh[8];
    const float mu  = s * (1.f / D_MODEL);
    const float var = ss * (1.f / D_MODEL) - mu * mu;
    const float inv = rsqrtf(var + 1e-5f);
    #pragma unroll
    for (int i = 0; i < 4; ++i) {
        int c = tid + i * 256;
        out[(size_t)row * D_MODEL + c] = (v[i] - mu) * inv * gam[c] + bet[c];
    }
}

// ---------------------------------------------------------------------------
extern "C" void kernel_launch(void* const* d_in, const int* in_sizes, int n_in,
                              void* d_out, int out_size, void* d_ws, size_t ws_size,
                              hipStream_t stream) {
    const float* tgt    = (const float*)d_in[0];
    const float* memory = (const float*)d_in[1];
    // d_in[2] = tgt_mask (causal handled analytically)
    const float* sa_qw = (const float*)d_in[3];  const float* sa_qb = (const float*)d_in[4];
    const float* sa_kw = (const float*)d_in[5];  const float* sa_kb = (const float*)d_in[6];
    const float* sa_vw = (const float*)d_in[7];  const float* sa_vb = (const float*)d_in[8];
    const float* sa_ow = (const float*)d_in[9];  const float* sa_ob = (const float*)d_in[10];
    const float* ca_qw = (const float*)d_in[11]; const float* ca_qb = (const float*)d_in[12];
    const float* ca_kw = (const float*)d_in[13]; const float* ca_kb = (const float*)d_in[14];
    const float* ca_vw = (const float*)d_in[15]; const float* ca_vb = (const float*)d_in[16];
    const float* ca_ow = (const float*)d_in[17]; const float* ca_ob = (const float*)d_in[18];
    const float* w1 = (const float*)d_in[19];    const float* b1 = (const float*)d_in[20];
    const float* w2 = (const float*)d_in[21];    const float* b2 = (const float*)d_in[22];
    const float* g1 = (const float*)d_in[23];    const float* be1 = (const float*)d_in[24];
    const float* g2 = (const float*)d_in[25];    const float* be2 = (const float*)d_in[26];
    const float* g3 = (const float*)d_in[27];    const float* be3 = (const float*)d_in[28];

    float* ws = (float*)d_ws;
    const size_t OFF = (size_t)MTOT * D_MODEL;   // 4M floats
    float* q   = ws;
    float* k   = ws + 1 * OFF;
    float* v   = ws + 2 * OFF;
    float* att = ws + 3 * OFF;
    float* tmp = ws + 4 * OFF;
    float* x1  = ws + 5 * OFF;
    float* x2  = ws + 6 * OFF;
    float* hid = ws;                              // reuse q..att (4*OFF) for FFN hidden

    const dim3 blk256(256), blk128(128);
    const dim3 gD(D_MODEL / 128, MTOT / 256);     // (8,16)
    const dim3 gF(D_FF / 128, MTOT / 256);        // (32,16)
    const dim3 gAttn(SEQ / 64, NHEAD, BATCH);

    // ---- self attention ----
    gemm_bias_kernel<<<gD, blk256, 0, stream>>>(tgt, sa_qw, sa_qb, q, MTOT, D_MODEL, D_MODEL, 0);
    gemm_bias_kernel<<<gD, blk256, 0, stream>>>(tgt, sa_kw, sa_kb, k, MTOT, D_MODEL, D_MODEL, 0);
    gemm_bias_kernel<<<gD, blk256, 0, stream>>>(tgt, sa_vw, sa_vb, v, MTOT, D_MODEL, D_MODEL, 0);
    flash_attn_kernel<<<gAttn, blk128, 0, stream>>>(q, k, v, att, 1);
    gemm_bias_kernel<<<gD, blk256, 0, stream>>>(att, sa_ow, sa_ob, tmp, MTOT, D_MODEL, D_MODEL, 0);
    add_ln_kernel<<<MTOT, blk256, 0, stream>>>(tgt, tmp, g1, be1, x1);

    // ---- cross attention ----
    gemm_bias_kernel<<<gD, blk256, 0, stream>>>(x1,     ca_qw, ca_qb, q, MTOT, D_MODEL, D_MODEL, 0);
    gemm_bias_kernel<<<gD, blk256, 0, stream>>>(memory, ca_kw, ca_kb, k, MTOT, D_MODEL, D_MODEL, 0);
    gemm_bias_kernel<<<gD, blk256, 0, stream>>>(memory, ca_vw, ca_vb, v, MTOT, D_MODEL, D_MODEL, 0);
    flash_attn_kernel<<<gAttn, blk128, 0, stream>>>(q, k, v, att, 0);
    gemm_bias_kernel<<<gD, blk256, 0, stream>>>(att, ca_ow, ca_ob, tmp, MTOT, D_MODEL, D_MODEL, 0);
    add_ln_kernel<<<MTOT, blk256, 0, stream>>>(x1, tmp, g2, be2, x2);

    // ---- FFN ----
    gemm_bias_kernel<<<gF, blk256, 0, stream>>>(x2, w1, b1, hid, MTOT, D_FF, D_MODEL, 1);
    gemm_bias_kernel<<<gD, blk256, 0, stream>>>(hid, w2, b2, tmp, MTOT, D_MODEL, D_FF, 0);
    add_ln_kernel<<<MTOT, blk256, 0, stream>>>(x2, tmp, g3, be3, (float*)d_out);
}
